// BilstmCRF_8151847927921
// MI455X (gfx1250) — matrix operand order repaired and checked by reading, not compiled, not run on hardware
//
#include <hip/hip_runtime.h>
#include <hip/hip_bf16.h>
#include <math.h>

// ---------------------------------------------------------------------------
// Model dims
#define BB   64
#define TT   512
#define DBERT 768
#define DD   1536          // 2*DBERT
#define CC   192
#define C4   768           // 4*C
#define HH   256
#define G4   1024          // 4*H
#define NLBL 20
#define BT   (BB*TT)       // 32768

typedef __attribute__((ext_vector_type(16))) __bf16 v16bf;
typedef __attribute__((ext_vector_type(8)))  __bf16 v8bf;
typedef __attribute__((ext_vector_type(8)))  float  v8f;
typedef unsigned int u32x4 __attribute__((ext_vector_type(4)));
typedef int          i32x8 __attribute__((ext_vector_type(8)));
typedef int          i32x4 __attribute__((ext_vector_type(4)));

#if defined(__gfx1250__) && __has_builtin(__builtin_amdgcn_tensor_load_to_lds) && \
    __has_builtin(__builtin_amdgcn_s_wait_tensorcnt)
#define HAVE_TDM 1
#else
#define HAVE_TDM 0
#endif

__device__ inline __bf16 f2bf(float f) {
  union { float f; unsigned u; } v; v.f = f;
  unsigned r = (v.u + 0x7FFFu + ((v.u >> 16) & 1u)) >> 16;   // RNE
  union { unsigned short s; __bf16 b; } o; o.s = (unsigned short)r;
  return o.b;
}

#if HAVE_TDM
// 2-D TDM load: tile_h rows x tile_w elements (dsz_code: 1=2B, 2=4B),
// row stride `stride` elements, into LDS offset lds_off.
// D# layout per CDNA5 ISA ch.8: group0 = {flags, lds_addr, gaddr_lo, gaddr_hi|type},
// group1 packs data_size, tensor_dim0/1, tile_dim0/1/2, tensor_dim0_stride.
// This toolchain's builtin takes 6 args (clang-23 form): g0,g1,g2,g3,g4,cpol.
__device__ inline void tdm_load_2d(unsigned lds_off, const void* gaddr,
                                   unsigned tile_w, unsigned tile_h,
                                   unsigned stride, unsigned dsz_code) {
  unsigned long long ga = (unsigned long long)(uintptr_t)gaddr;
  u32x4 g0;
  g0[0] = 1u;                                            // count=1, is_restore=0
  g0[1] = lds_off;                                       // LDS byte address
  g0[2] = (unsigned)ga;                                  // global_addr[31:0]
  g0[3] = (unsigned)((ga >> 32) & 0x01FFFFFFull) | 0x80000000u; // [56:32] | type=2
  i32x8 g1;
  g1[0] = (int)(dsz_code << 16);   // workgroup_mask=0, data_size
  g1[1] = (int)(tile_w << 16);     // tensor_dim0[15:0] (== tile_w, in-bounds)
  g1[2] = (int)(tile_h << 16);     // tensor_dim1[15:0]
  g1[3] = (int)(tile_w << 16);     // tile_dim0
  g1[4] = (int)tile_h;             // tile_dim1 (tile_dim2 = 0)
  g1[5] = (int)stride;             // tensor_dim0_stride[31:0]
  g1[6] = 0;
  g1[7] = 0;                       // tensor_dim1_stride = 0 (2-D)
  i32x4 z4 = {0, 0, 0, 0};
  i32x8 z8 = {0, 0, 0, 0, 0, 0, 0, 0};
  __builtin_amdgcn_tensor_load_to_lds(g0, g1, z4, z4, z8, 0);
}
#endif

// ---------------------------------------------------------------------------
// Kernel 1: concat(hid_a,hid_b) -> LayerNorm(D=1536) -> bf16
__global__ __launch_bounds__(256)
void ln1_kernel(const float* __restrict__ ha, const float* __restrict__ hb,
                const float* __restrict__ g, const float* __restrict__ bta,
                __bf16* __restrict__ out) {
  const int row = blockIdx.x;
  const int tid = threadIdx.x;
  float x[6];
  float s = 0.f;
#pragma unroll
  for (int i = 0; i < 6; ++i) {
    int k = tid + i * 256;
    float v = (k < DBERT) ? ha[(size_t)row * DBERT + k]
                          : hb[(size_t)row * DBERT + (k - DBERT)];
    x[i] = v; s += v;
  }
  __shared__ float red[256];
  red[tid] = s; __syncthreads();
  for (int off = 128; off > 0; off >>= 1) {
    if (tid < off) red[tid] += red[tid + off];
    __syncthreads();
  }
  const float mean = red[0] * (1.f / DD);
  __syncthreads();
  float sq = 0.f;
#pragma unroll
  for (int i = 0; i < 6; ++i) { float d = x[i] - mean; sq += d * d; }
  red[tid] = sq; __syncthreads();
  for (int off = 128; off > 0; off >>= 1) {
    if (tid < off) red[tid] += red[tid + off];
    __syncthreads();
  }
  const float rs = rsqrtf(red[0] * (1.f / DD) + 1e-5f);
#pragma unroll
  for (int i = 0; i < 6; ++i) {
    int k = tid + i * 256;
    out[(size_t)row * DD + k] = f2bf((x[i] - mean) * rs * g[k] + bta[k]);
  }
}

// ---------------------------------------------------------------------------
// Kernel 2: 4 conv1d (widths 1..4, SAME) as tap-GEMMs via WMMA bf16 + ReLU.
// grid = (BT/16, 12), block = 128 (4 waves). Each block: 16 rows x 64 cols of
// one conv; weight K-slab (32x64 f32) staged into LDS by the TDM.
__global__ __launch_bounds__(128)
void conv_wmma_kernel(const __bf16* __restrict__ xln,
                      const float* __restrict__ w1, const float* __restrict__ b1,
                      const float* __restrict__ w2, const float* __restrict__ b2,
                      const float* __restrict__ w3, const float* __restrict__ b3,
                      const float* __restrict__ w4, const float* __restrict__ b4,
                      float* __restrict__ conv) {
  const int tid  = threadIdx.x;
  const int lane = tid & 31;
  const int wv   = tid >> 5;               // 0..3
  const int gy   = blockIdx.y;             // 0..11
  const int ci   = gy / 3;                 // which conv (0..3)
  const int colbase = (gy % 3) * 64;       // column group within that conv
  const int width = ci + 1;
  const int pad   = (width - 1) / 2;       // SAME: pad_low = (k-1)//2
  const float* W  = (ci == 0) ? w1 : (ci == 1) ? w2 : (ci == 2) ? w3 : w4;
  const float* Bv = (ci == 0) ? b1 : (ci == 1) ? b2 : (ci == 2) ? b3 : b4;

  const int Mbase = blockIdx.x * 16;
  const int b     = Mbase / TT;                 // tile never crosses batch
  const int t0    = (Mbase % TT) + (lane & 15); // per-lane A-row time
  __shared__ float wlds[32 * 64];               // 8 KB weight slab
  v8f acc = {0.f, 0.f, 0.f, 0.f, 0.f, 0.f, 0.f, 0.f};
  const int nloc = wv * 16 + (lane & 15);       // local column 0..63

  for (int j = 0; j < width; ++j) {
    const int ta = t0 + (j - pad);
    const bool valid = (ta >= 0) && (ta < TT);
    const __bf16* arow = xln + ((size_t)(b * TT + (valid ? ta : 0))) * DD;
    const float* wj = W + (size_t)j * DD * CC + colbase;
    for (int kk = 0; kk < DD; kk += 32) {
      __syncthreads();   // previous slab fully consumed
#if HAVE_TDM
      if (wv == 0) {
        tdm_load_2d(0u, wj + (size_t)kk * CC, 64u, 32u, (unsigned)CC, 2u);
        __builtin_amdgcn_s_wait_tensorcnt(0);
      }
#else
      for (int i = tid; i < 32 * 64; i += 128)
        wlds[i] = wj[(size_t)(kk + (i >> 6)) * CC + (i & 63)];
#endif
      __syncthreads();
      if (valid) __builtin_prefetch(arow + kk + 64, 0, 3);
      v16bf a;
      if (valid) {
        const __bf16* p = arow + kk + ((lane & 16) ? 16 : 0);
        v8bf lo = *(const v8bf*)p;
        v8bf hi = *(const v8bf*)(p + 8);
#pragma unroll
        for (int e = 0; e < 8; ++e) { a[e] = lo[e]; a[e + 8] = hi[e]; }
      } else {
#pragma unroll
        for (int e = 0; e < 16; ++e) a[e] = f2bf(0.f);
      }
      v16bf bf;
      const int kb = (lane & 16) ? 16 : 0;
#pragma unroll
      for (int e = 0; e < 16; ++e) bf[e] = f2bf(wlds[(kb + e) * 64 + nloc]);
      acc = __builtin_amdgcn_wmma_f32_16x16x32_bf16(false, a, false, bf,
                                                    (short)0, acc, false, false);
    }
  }
  const int col = ci * CC + colbase + nloc;     // channel in concat output
  const float bv = Bv[colbase + nloc];
#pragma unroll
  for (int r = 0; r < 8; ++r) {
    int row = Mbase + r + ((lane & 16) ? 8 : 0);
    float v = acc[r] + bv;
    conv[(size_t)row * C4 + col] = v > 0.f ? v : 0.f;
  }
}

// ---------------------------------------------------------------------------
// Kernel 3: LayerNorm(768) over conv output -> bf16
__global__ __launch_bounds__(256)
void ln2_kernel(const float* __restrict__ conv,
                const float* __restrict__ g, const float* __restrict__ bta,
                __bf16* __restrict__ out) {
  const int row = blockIdx.x;
  const int tid = threadIdx.x;
  float x[3];
  float s = 0.f;
#pragma unroll
  for (int i = 0; i < 3; ++i) {
    int k = tid + i * 256;
    float v = conv[(size_t)row * C4 + k];
    x[i] = v; s += v;
  }
  __shared__ float red[256];
  red[tid] = s; __syncthreads();
  for (int off = 128; off > 0; off >>= 1) {
    if (tid < off) red[tid] += red[tid + off];
    __syncthreads();
  }
  const float mean = red[0] * (1.f / C4);
  __syncthreads();
  float sq = 0.f;
#pragma unroll
  for (int i = 0; i < 3; ++i) { float d = x[i] - mean; sq += d * d; }
  red[tid] = sq; __syncthreads();
  for (int off = 128; off > 0; off >>= 1) {
    if (tid < off) red[tid] += red[tid + off];
    __syncthreads();
  }
  const float rs = rsqrtf(red[0] * (1.f / C4) + 1e-5f);
#pragma unroll
  for (int i = 0; i < 3; ++i) {
    int k = tid + i * 256;
    out[(size_t)row * C4 + k] = f2bf((x[i] - mean) * rs * g[k] + bta[k]);
  }
}

// ---------------------------------------------------------------------------
// Kernel 4: convert wx_f, wx_b, wh_f, wh_b to bf16 (packed sequentially)
__global__ __launch_bounds__(256)
void wprep_kernel(const float* __restrict__ wxf, const float* __restrict__ wxb,
                  const float* __restrict__ whf, const float* __restrict__ whb,
                  __bf16* __restrict__ out) {
  const int NX = C4 * G4;   // 786432
  const int NH = HH * G4;   // 262144
  int i = blockIdx.x * 256 + threadIdx.x;
  if (i < NX)                   out[i] = f2bf(wxf[i]);
  else if (i < 2 * NX)          out[i] = f2bf(wxb[i - NX]);
  else if (i < 2 * NX + NH)     out[i] = f2bf(whf[i - 2 * NX]);
  else if (i < 2 * NX + 2 * NH) out[i] = f2bf(whb[i - 2 * NX - NH]);
}

// ---------------------------------------------------------------------------
// Kernel 5: xproj_d = conv_ln @ wx_d + bias_d (WMMA, K=768).
// grid = (BT/16, 16, 2), block = 128 (4 waves / 64 cols); bf16 weight slab
// (32x64) staged into LDS by the TDM each K step.
__global__ __launch_bounds__(128)
void xproj_wmma_kernel(const __bf16* __restrict__ cln,
                       const __bf16* __restrict__ wx_all,
                       const float* __restrict__ bf_, const float* __restrict__ bb_,
                       float* __restrict__ xpf, float* __restrict__ xpb) {
  const int tid  = threadIdx.x;
  const int lane = tid & 31;
  const int wv   = tid >> 5;
  const int Mbase   = blockIdx.x * 16;
  const int colbase = blockIdx.y * 64;
  const int dir     = blockIdx.z;
  const __bf16* wx  = wx_all + (size_t)dir * C4 * G4 + colbase;
  const float* bias = dir ? bb_ : bf_;
  float* out        = dir ? xpb : xpf;

  __shared__ __bf16 wlds[32 * 64];   // 4 KB bf16 weight slab
  v8f acc = {0.f, 0.f, 0.f, 0.f, 0.f, 0.f, 0.f, 0.f};
  const __bf16* arow = cln + (size_t)(Mbase + (lane & 15)) * C4;
  const int nloc = wv * 16 + (lane & 15);

  for (int kk = 0; kk < C4; kk += 32) {
    __syncthreads();
#if HAVE_TDM
    if (wv == 0) {
      tdm_load_2d(0u, wx + (size_t)kk * G4, 64u, 32u, (unsigned)G4, 1u);
      __builtin_amdgcn_s_wait_tensorcnt(0);
    }
#else
    for (int i = tid; i < 32 * 64; i += 128)
      wlds[i] = wx[(size_t)(kk + (i >> 6)) * G4 + (i & 63)];
#endif
    __syncthreads();
    __builtin_prefetch(arow + kk + 64, 0, 3);
    v16bf a;
    const __bf16* p = arow + kk + ((lane & 16) ? 16 : 0);
    v8bf lo = *(const v8bf*)p;
    v8bf hi = *(const v8bf*)(p + 8);
#pragma unroll
    for (int e = 0; e < 8; ++e) { a[e] = lo[e]; a[e + 8] = hi[e]; }
    v16bf bf;
    const int kb = (lane & 16) ? 16 : 0;
#pragma unroll
    for (int e = 0; e < 16; ++e) bf[e] = wlds[(kb + e) * 64 + nloc];
    acc = __builtin_amdgcn_wmma_f32_16x16x32_bf16(false, a, false, bf,
                                                  (short)0, acc, false, false);
  }
  const int n = colbase + nloc;
  const float bv = bias[n];
#pragma unroll
  for (int r = 0; r < 8; ++r)
    out[(size_t)(Mbase + r + ((lane & 16) ? 8 : 0)) * G4 + n] = acc[r] + bv;
}

// ---------------------------------------------------------------------------
// Kernel 6: LSTM recurrence. grid=2 (direction), block=1024 (32 waves).
// Per step: waves WMMA h@Wh (A = bf16 h in LDS, B = bf16 Wh in L2),
// z staged through L2 scratch, pointwise gate phase, h rewritten to LDS.
__global__ __launch_bounds__(1024)
void lstm_kernel(const float* __restrict__ xpf, const float* __restrict__ xpb,
                 const __bf16* __restrict__ wh_all,
                 const int* __restrict__ am,
                 float* __restrict__ hf, float* __restrict__ hbk,
                 float* __restrict__ scratch) {
  const int dir = blockIdx.x;
  const float*  xp  = dir ? xpb : xpf;
  const __bf16* wh  = wh_all + (size_t)dir * HH * G4;
  float* hout       = dir ? hbk : hf;
  float* zbuf = scratch + (size_t)dir * (BB * G4 + 3 * BB * HH);
  float* cbuf = zbuf + BB * G4;
  float* hsb  = cbuf + BB * HH;     // fp32 h carry
  float* opb  = hsb + BB * HH;      // "out" carry

  const int tid  = threadIdx.x;
  const int lane = tid & 31;
  const int wv   = tid >> 5;
  __shared__ __bf16 hsh[BB * HH];   // 32 KB

  const __bf16 bz = f2bf(0.f);
  for (int i = tid; i < BB * HH; i += 1024) {
    hsh[i] = bz; cbuf[i] = 0.f; hsb[i] = 0.f; opb[i] = 0.f;
  }
  __threadfence();
  __syncthreads();

  const int Ncol = wv * 32;
  for (int step = 0; step < TT; ++step) {
    const int t = dir ? (TT - 1 - step) : step;
    v8f acc[2][4];
#pragma unroll
    for (int nt = 0; nt < 2; ++nt)
#pragma unroll
      for (int mt = 0; mt < 4; ++mt)
#pragma unroll
        for (int r = 0; r < 8; ++r) acc[nt][mt][r] = 0.f;

    for (int kk = 0; kk < HH; kk += 32) {
      v16bf afr[4];
#pragma unroll
      for (int mt = 0; mt < 4; ++mt) {
        const __bf16* p = &hsh[(mt * 16 + (lane & 15)) * HH + kk + ((lane & 16) ? 16 : 0)];
        v8bf lo = *(const v8bf*)p;
        v8bf hi = *(const v8bf*)(p + 8);
#pragma unroll
        for (int e = 0; e < 8; ++e) { afr[mt][e] = lo[e]; afr[mt][e + 8] = hi[e]; }
      }
      v16bf bfr[2];
#pragma unroll
      for (int nt = 0; nt < 2; ++nt) {
        const int n  = Ncol + nt * 16 + (lane & 15);
        const int kb = kk + ((lane & 16) ? 16 : 0);
#pragma unroll
        for (int e = 0; e < 16; ++e) bfr[nt][e] = wh[(size_t)(kb + e) * G4 + n];
      }
#pragma unroll
      for (int nt = 0; nt < 2; ++nt)
#pragma unroll
        for (int mt = 0; mt < 4; ++mt)
          acc[nt][mt] = __builtin_amdgcn_wmma_f32_16x16x32_bf16(
              false, afr[mt], false, bfr[nt], (short)0, acc[nt][mt], false, false);
    }
#pragma unroll
    for (int nt = 0; nt < 2; ++nt) {
      const int n = Ncol + nt * 16 + (lane & 15);
#pragma unroll
      for (int mt = 0; mt < 4; ++mt)
#pragma unroll
        for (int r = 0; r < 8; ++r) {
          const int row = mt * 16 + r + ((lane & 16) ? 8 : 0);   // batch index
          zbuf[row * G4 + n] = acc[nt][mt][r] + xp[((size_t)row * TT + t) * G4 + n];
        }
    }
    __threadfence();
    __syncthreads();
    for (int u = 0; u < 16; ++u) {
      const int idx = tid + u * 1024;     // 0..16383 = r*256+j
      const int r = idx >> 8, j = idx & 255;
      const float zi = zbuf[r * G4 + j];
      const float zf = zbuf[r * G4 + HH + j];
      const float zg = zbuf[r * G4 + 2 * HH + j];
      const float zo = zbuf[r * G4 + 3 * HH + j];
      const float ii = 1.f / (1.f + __expf(-zi));
      const float ff = 1.f / (1.f + __expf(-zf));
      const float oo = 1.f / (1.f + __expf(-zo));
      const float gg = tanhf(zg);
      const float cold = cbuf[idx];
      const float cnew = ff * cold + ii * gg;
      const float hnew = oo * tanhf(cnew);
      const bool  m = am[r * TT + t] != 0;
      const float h2 = m ? hnew : hsb[idx];
      const float c2 = m ? cnew : cold;
      const float ov = m ? hnew : opb[idx];
      cbuf[idx] = c2; hsb[idx] = h2; opb[idx] = ov;
      hsh[idx] = f2bf(h2);
      hout[((size_t)r * TT + t) * HH + j] = ov;
    }
    __threadfence();
    __syncthreads();
  }
}

// ---------------------------------------------------------------------------
// Kernel 7: logits = [hf|hb] @ wd + bd
__global__ __launch_bounds__(256)
void logits_kernel(const float* __restrict__ hf, const float* __restrict__ hb,
                   const float* __restrict__ wd, const float* __restrict__ bd,
                   float* __restrict__ logits) {
  const int idx = blockIdx.x * 256 + threadIdx.x;
  if (idx >= BT * NLBL) return;
  const int bt = idx / NLBL, l = idx % NLBL;
  float s = bd[l];
  const float* h0 = hf + (size_t)bt * HH;
  const float* h1 = hb + (size_t)bt * HH;
  for (int k = 0; k < HH; ++k) s += h0[k] * wd[k * NLBL + l];
  for (int k = 0; k < HH; ++k) s += h1[k] * wd[(HH + k) * NLBL + l];
  logits[(size_t)bt * NLBL + l] = s;
}

// ---------------------------------------------------------------------------
// Kernel 8: CRF log-likelihood. One wave per batch element; lane j holds alpha[j].
__global__ __launch_bounds__(32)
void crf_kernel(const float* __restrict__ logits, const int* __restrict__ inputs,
                const int* __restrict__ targets, const float* __restrict__ trans,
                float* __restrict__ out) {
  const int b = blockIdx.x;
  const int lane = threadIdx.x;
  int cnt = 0;
  for (int t = lane; t < TT; t += 32) cnt += (inputs[b * TT + t] != 0) ? 1 : 0;
#pragma unroll
  for (int off = 16; off > 0; off >>= 1) cnt += __shfl_xor(cnt, off, 32);
  const int seqlen = cnt;
  float un = 0.f;
  for (int t = lane; t < TT; t += 32)
    if (t < seqlen) {
      int tg = targets[b * TT + t];
      un += logits[((size_t)b * TT + t) * NLBL + tg];
    }
#pragma unroll
  for (int off = 16; off > 0; off >>= 1) un += __shfl_xor(un, off, 32);
  float bi = 0.f;
  for (int t = 1 + lane; t < TT; t += 32)
    if (t < seqlen) {
      int p0 = targets[b * TT + t - 1], p1 = targets[b * TT + t];
      bi += trans[p0 * NLBL + p1];
    }
#pragma unroll
  for (int off = 16; off > 0; off >>= 1) bi += __shfl_xor(bi, off, 32);
  const int j = (lane < NLBL) ? lane : 0;
  float alpha = (lane < NLBL) ? logits[((size_t)b * TT) * NLBL + lane] : -1e30f;
  for (int t = 1; t < TT; ++t) {
    const float emit = logits[((size_t)b * TT + t) * NLBL + j];
    float vs[NLBL];
    float mx = -1e30f;
#pragma unroll
    for (int i = 0; i < NLBL; ++i) {
      float ai = __shfl(alpha, i, 32);
      float v = ai + trans[i * NLBL + j];
      vs[i] = v;
      mx = v > mx ? v : mx;
    }
    float sum = 0.f;
#pragma unroll
    for (int i = 0; i < NLBL; ++i) sum += __expf(vs[i] - mx);
    const float nw = mx + __logf(sum) + emit;
    if (t < seqlen && lane < NLBL) alpha = nw;
  }
  float mx = alpha;
#pragma unroll
  for (int off = 16; off > 0; off >>= 1) {
    float o = __shfl_xor(mx, off, 32);
    mx = o > mx ? o : mx;
  }
  float s = (lane < NLBL) ? __expf(alpha - mx) : 0.f;
#pragma unroll
  for (int off = 16; off > 0; off >>= 1) s += __shfl_xor(s, off, 32);
  const float lnorm = mx + __logf(s);
  if (lane == 0) out[b] = un + bi - lnorm;
}

// ---------------------------------------------------------------------------
extern "C" void kernel_launch(void* const* d_in, const int* in_sizes, int n_in,
                              void* d_out, int out_size, void* d_ws, size_t ws_size,
                              hipStream_t stream) {
  (void)in_sizes; (void)n_in; (void)out_size; (void)ws_size;
  const int*   inputs = (const int*)d_in[0];
  const int*   am     = (const int*)d_in[1];
  const int*   tgt    = (const int*)d_in[2];
  const float* hid_a  = (const float*)d_in[3];
  const float* hid_b  = (const float*)d_in[4];
  const float* ln1_g  = (const float*)d_in[5];
  const float* ln1_b  = (const float*)d_in[6];
  const float* w1 = (const float*)d_in[7];  const float* b1 = (const float*)d_in[8];
  const float* w2 = (const float*)d_in[9];  const float* b2 = (const float*)d_in[10];
  const float* w3 = (const float*)d_in[11]; const float* b3 = (const float*)d_in[12];
  const float* w4 = (const float*)d_in[13]; const float* b4 = (const float*)d_in[14];
  const float* ln2_g = (const float*)d_in[15];
  const float* ln2_b = (const float*)d_in[16];
  const float* wx_f = (const float*)d_in[17];
  const float* wh_f = (const float*)d_in[18];
  const float* bf_  = (const float*)d_in[19];
  const float* wx_b = (const float*)d_in[20];
  const float* wh_b = (const float*)d_in[21];
  const float* bb_  = (const float*)d_in[22];
  const float* wd   = (const float*)d_in[23];
  const float* bd   = (const float*)d_in[24];
  const float* trans = (const float*)d_in[25];
  float* outp = (float*)d_out;

  char* ws = (char*)d_ws;
  const size_t XLN_SZ  = (size_t)BT * DD * 2;   // 96 MB bf16
  const size_t CONV_SZ = (size_t)BT * C4 * 4;   // 96 MB f32
  const size_t CLN_SZ  = (size_t)BT * C4 * 2;   // 48 MB bf16
  const size_t XP_SZ   = (size_t)BT * G4 * 4;   // 128 MB f32
  const size_t XLN_OFF  = 0;
  const size_t CONV_OFF = XLN_OFF + XLN_SZ;
  const size_t CLN_OFF  = CONV_OFF + CONV_SZ;
  const size_t XPF_OFF  = CLN_OFF + CLN_SZ;
  const size_t XPB_OFF  = XPF_OFF + XP_SZ;
  const size_t WBF_OFF  = XPB_OFF + XP_SZ;

  __bf16* xln   = (__bf16*)(ws + XLN_OFF);
  float*  conv  = (float*)(ws + CONV_OFF);
  __bf16* cln   = (__bf16*)(ws + CLN_OFF);
  float*  xpf   = (float*)(ws + XPF_OFF);
  float*  xpb   = (float*)(ws + XPB_OFF);
  __bf16* wbf   = (__bf16*)(ws + WBF_OFF);
  __bf16* whbf  = wbf + 2 * (size_t)C4 * G4;
  // aliases over dead regions:
  float* hf = (float*)(ws + CONV_OFF);                          // 32 MB
  float* hb = (float*)(ws + CONV_OFF + (size_t)BT * HH * 4);    // 32 MB
  float* lstm_scr = (float*)(ws + XLN_OFF);                     // < 1 MB
  float* logits   = (float*)(ws + XLN_OFF + (size_t)4 * 1024 * 1024); // 2.5 MB

  ln1_kernel<<<BT, 256, 0, stream>>>(hid_a, hid_b, ln1_g, ln1_b, xln);
  conv_wmma_kernel<<<dim3(BT / 16, 12), 128, 0, stream>>>(
      xln, w1, b1, w2, b2, w3, b3, w4, b4, conv);
  ln2_kernel<<<BT, 256, 0, stream>>>(conv, ln2_g, ln2_b, cln);
  {
    const int NTOT = 2 * C4 * G4 + 2 * HH * G4;
    wprep_kernel<<<(NTOT + 255) / 256, 256, 0, stream>>>(wx_f, wx_b, wh_f, wh_b, wbf);
  }
  xproj_wmma_kernel<<<dim3(BT / 16, 16, 2), 128, 0, stream>>>(
      cln, wbf, bf_, bb_, xpf, xpb);
  lstm_kernel<<<2, 1024, 0, stream>>>(xpf, xpb, whbf, am, hf, hb, lstm_scr);
  logits_kernel<<<(BT * NLBL + 255) / 256, 256, 0, stream>>>(hf, hb, wd, bd, logits);
  crf_kernel<<<BB, 32, 0, stream>>>(logits, inputs, tgt, trans, outp);
}
